// SpatioTemporalGraphSAGERaw_56401510531288
// MI455X (gfx1250) — matrix operand order repaired and checked by reading, not compiled, not run on hardware
//
#include <hip/hip_runtime.h>
#include <hip/hip_bf16.h>
#include <stdint.h>

typedef __attribute__((ext_vector_type(16))) __bf16       v16bf;
typedef __attribute__((ext_vector_type(8)))  float        v8f;
typedef __attribute__((ext_vector_type(4)))  unsigned int v4u;
typedef __attribute__((ext_vector_type(8)))  unsigned int v8u;

#define T_FR   30
#define JNT    17
#define NPS    510                 // nodes per sample (T*J)
#define BATCH  512
#define NTOT   (BATCH * NPS)       // 261120 rows total
#define BN_EPS 1e-5f

// ---------- bf16 helpers (round-to-nearest-even) ----------
__device__ __forceinline__ unsigned short f2bf(float f) {
  unsigned u = __float_as_uint(f);
  u += 0x7FFFu + ((u >> 16) & 1u);
  return (unsigned short)(u >> 16);
}
__device__ __forceinline__ float bf2f(unsigned short h) {
  return __uint_as_float(((unsigned)h) << 16);
}

// ---------- fixed COCO skeleton adjacency (spatial neighbors per joint) ----------
__constant__ int NBR_CNT[17] = {2,2,2,1,1,3,3,2,2,1,1,3,3,2,2,1,1};
__constant__ int NBR[17][3]  = {{1,2,0},{0,3,0},{0,4,0},{1,0,0},{2,0,0},
                                {6,7,11},{5,8,12},{5,9,0},{6,10,0},{7,0,0},{8,0,0},
                                {5,12,13},{6,11,14},{11,15,0},{12,16,0},{13,0,0},{14,0,0}};

// ---------- WMMA fragment loads ----------
// A (16x32 bf16, row-major src): lane holds row (lane&15); lanes<16 take K 0-15,
// lanes>=16 take K 16-31  -> two 16B chunks at p and p+8 halves.
__device__ __forceinline__ v16bf load_a_frag(const unsigned short* p) {
  v4u lo = *(const v4u*)(p);
  v4u hi = *(const v4u*)(p + 8);
  v8u r; r[0]=lo[0]; r[1]=lo[1]; r[2]=lo[2]; r[3]=lo[3];
         r[4]=hi[0]; r[5]=hi[1]; r[6]=hi[2]; r[7]=hi[3];
  return __builtin_bit_cast(v16bf, r);
}
// B (32x16 bf16) from transposed weight WT[dout][din]: lane holds column (lane&15),
// K 0-15 (lanes<16) or 16-31 (lanes>=16), contiguous -> one 32B load.
__device__ __forceinline__ v16bf load_b_frag(const unsigned short* p) {
  return __builtin_bit_cast(v16bf, *(const v8u*)p);
}

// ---------- fused SAGE GEMM: y = agg@Wl + x@Wr + bl ; xres = x@Wres + bres ----------
extern "C" __global__ __launch_bounds__(256)
void sage_gemm(const unsigned short* __restrict__ aggB,
               const unsigned short* __restrict__ xB,
               const unsigned short* __restrict__ WlT,
               const unsigned short* __restrict__ WrT,
               const unsigned short* __restrict__ WresT,
               const float* __restrict__ bl, const float* __restrict__ bres,
               float* __restrict__ y, float* __restrict__ xres,
               int din, int dout)
{
  const int wave    = threadIdx.x >> 5;
  const int lane    = threadIdx.x & 31;
  const int halfSel = lane >> 4;
  const int l15     = lane & 15;
  const int tileM   = (blockIdx.x * 8 + wave) << 4;
  const int tileN   = blockIdx.y << 4;

  const size_t rowOff = (size_t)(tileM + l15) * din;
  const unsigned short* aggRow = aggB + rowOff;
  const unsigned short* xRow   = xB   + rowOff;
  const int nCol = tileN + l15;
  const size_t wOff = (size_t)nCol * din + halfSel * 16;
  const unsigned short* wl = WlT   + wOff;
  const unsigned short* wr = WrT   + wOff;
  const unsigned short* wq = WresT + wOff;
  const int aOff = halfSel * 16;

  v8f accY = {0.f,0.f,0.f,0.f,0.f,0.f,0.f,0.f};
  v8f accR = accY;

  for (int k0 = 0; k0 < din; k0 += 32) {
    __builtin_prefetch(aggRow + k0 + 64, 0, 1);
    __builtin_prefetch(xRow   + k0 + 64, 0, 1);
    v16bf aAgg = load_a_frag(aggRow + k0 + aOff);
    v16bf aX   = load_a_frag(xRow   + k0 + aOff);
    v16bf bL   = load_b_frag(wl + k0);
    v16bf bR   = load_b_frag(wr + k0);
    v16bf bQ   = load_b_frag(wq + k0);
    accY = __builtin_amdgcn_wmma_f32_16x16x32_bf16(false, aAgg, false, bL, (short)0, accY, false, false);
    accY = __builtin_amdgcn_wmma_f32_16x16x32_bf16(false, aX,   false, bR, (short)0, accY, false, false);
    accR = __builtin_amdgcn_wmma_f32_16x16x32_bf16(false, aX,   false, bQ, (short)0, accR, false, false);
  }

  const float by = bl[nCol];
  const float br = bres[nCol];
  const int mBase = tileM + (halfSel << 3);
#pragma unroll
  for (int r = 0; r < 8; ++r) {
    size_t o = (size_t)(mBase + r) * dout + nCol;
    y[o]    = accY[r] + by;
    xres[o] = accR[r] + br;
  }
}

// ---------- weight convert + transpose: W[din][dout] f32 -> WT[dout][din] bf16 ----------
extern "C" __global__ void wconv_t(const float* __restrict__ W,
                                   unsigned short* __restrict__ WT, int din, int dout)
{
  int idx = blockIdx.x * blockDim.x + threadIdx.x;
  if (idx >= din * dout) return;
  int o = idx / din, i = idx - o * din;
  WT[idx] = f2bf(W[(size_t)i * dout + o]);
}

// ---------- input projection: x0_bf16 = x_seq @ proj_w + proj_b ----------
extern "C" __global__ __launch_bounds__(256)
void proj_kernel(const float* __restrict__ xs, const float* __restrict__ w,
                 const float* __restrict__ b, unsigned short* __restrict__ xb)
{
  size_t idx = (size_t)blockIdx.x * blockDim.x + threadIdx.x;
  if (idx >= (size_t)NTOT * 64) return;
  size_t n = idx >> 6; int c = (int)(idx & 63);
  const float* xr = xs + n * 3;
  float v = xr[0]*w[c] + xr[1]*w[64+c] + xr[2]*w[128+c] + b[c];
  xb[idx] = f2bf(v);
}

// ---------- gather-mean aggregation using the fixed skeleton graph ----------
extern "C" __global__ __launch_bounds__(256)
void agg_mean(const unsigned short* __restrict__ xb,
              unsigned short* __restrict__ agg, int shift)
{
  size_t idx = (size_t)blockIdx.x * blockDim.x + threadIdx.x;
  size_t total = (size_t)NTOT << shift;
  if (idx >= total) return;
  int din = 1 << shift;
  size_t n = idx >> shift;
  int c = (int)(idx & (din - 1));
  int local = (int)(n % NPS);
  int t = local / JNT;
  int j = local - t * JNT;
  float s = 0.f;
  int cnt = NBR_CNT[j];
  size_t nodeBase = n - j;   // node id of joint 0 at this (sample, frame)
  for (int k = 0; k < cnt; ++k)
    s += bf2f(xb[((nodeBase + NBR[j][k]) << shift) + c]);
  int deg = cnt;
  if (t > 0)        { s += bf2f(xb[((n - JNT) << shift) + c]); ++deg; }
  if (t < T_FR - 1) { s += bf2f(xb[((n + JNT) << shift) + c]); ++deg; }
  agg[idx] = f2bf(s / (float)deg);
}

// ---------- BatchNorm stats pass 1: per-stripe column partial sums (deterministic) ----------
extern "C" __global__ __launch_bounds__(256)
void col_stats_partial(const float* __restrict__ y, float* __restrict__ partial, int dout)
{
  __shared__ float s1[256], s2[256];
  const int tx = threadIdx.x & 15, ty = threadIdx.x >> 4;
  const int col = blockIdx.x * 16 + tx;
  const int row0 = blockIdx.y * 1024;
  float s = 0.f, ss = 0.f;
  for (int r = row0 + ty; r < row0 + 1024; r += 16) {
    float v = y[(size_t)r * dout + col];
    s += v; ss += v * v;
  }
  s1[threadIdx.x] = s; s2[threadIdx.x] = ss;
  __syncthreads();
  for (int off = 128; off >= 16; off >>= 1) {
    if (threadIdx.x < off) { s1[threadIdx.x] += s1[threadIdx.x+off];
                             s2[threadIdx.x] += s2[threadIdx.x+off]; }
    __syncthreads();
  }
  if (ty == 0) {
    float* dst = partial + (size_t)blockIdx.y * 2 * dout;
    dst[col]        = s1[tx];
    dst[dout + col] = s2[tx];
  }
}

// ---------- BatchNorm stats pass 2: reduce stripes in fixed order ----------
extern "C" __global__ void col_stats_reduce(const float* __restrict__ partial,
                                            float* __restrict__ stats, int dout, int nPart)
{
  int i = blockIdx.x * blockDim.x + threadIdx.x;
  if (i >= 2 * dout) return;
  float s = 0.f;
  for (int p = 0; p < nPart; ++p) s += partial[(size_t)p * 2 * dout + i];
  stats[i] = s;
}

// ---------- finalize: x = relu(BN(y)) + xres  (in-place over y; also emit bf16 copy) ----------
extern "C" __global__ __launch_bounds__(256)
void bn_relu_res(float* __restrict__ y, const float* __restrict__ xres,
                 const float* __restrict__ stats, const float* __restrict__ g,
                 const float* __restrict__ b, unsigned short* __restrict__ xb16,
                 int dout, int shift)
{
  size_t idx = (size_t)blockIdx.x * blockDim.x + threadIdx.x;
  size_t total = (size_t)NTOT << shift;
  if (idx >= total) return;
  int c = (int)(idx & (dout - 1));
  const float invN = 1.0f / (float)NTOT;
  float mu   = stats[c] * invN;
  float var  = stats[dout + c] * invN - mu * mu;
  float rstd = rsqrtf(var + BN_EPS);
  float v = (y[idx] - mu) * rstd * g[c] + b[c];
  v = fmaxf(v, 0.f) + xres[idx];
  y[idx] = v;
  xb16[idx] = f2bf(v);
}

// ---------- multi-head attention pooling (one block per sample) ----------
extern "C" __global__ __launch_bounds__(128)
void attn_pool(const float* __restrict__ x, const float* __restrict__ aw,
               const float* __restrict__ ab, float* __restrict__ pooled)
{
  __shared__ float sc[NPS * 4];
  const int b = blockIdx.x, tid = threadIdx.x;
  const float* xb = x + (size_t)b * NPS * 128;
  for (int n = tid; n < NPS; n += 128) {
    const float* xr = xb + (size_t)n * 128;
    for (int h = 0; h < 4; ++h) {
      const float* wh = aw + h * 128;
      float d = 0.f;
      for (int k = 0; k < 128; ++k) d += xr[k] * wh[k];
      sc[n * 4 + h] = d + ab[h];
    }
  }
  __syncthreads();
  if (tid < 4) {
    float mx = -3.4e38f;
    for (int n = 0; n < NPS; ++n) mx = fmaxf(mx, sc[n*4+tid]);
    float s = 0.f;
    for (int n = 0; n < NPS; ++n) { float e = __expf(sc[n*4+tid] - mx); sc[n*4+tid] = e; s += e; }
    float inv = 1.f / s;
    for (int n = 0; n < NPS; ++n) sc[n*4+tid] *= inv;
  }
  __syncthreads();
  float a0=0.f, a1=0.f, a2=0.f, a3=0.f;
  for (int n = 0; n < NPS; ++n) {
    float xv = xb[(size_t)n * 128 + tid];
    a0 += sc[n*4+0]*xv; a1 += sc[n*4+1]*xv; a2 += sc[n*4+2]*xv; a3 += sc[n*4+3]*xv;
  }
  float* pb = pooled + (size_t)b * 512;
  pb[tid] = a0; pb[128+tid] = a1; pb[256+tid] = a2; pb[384+tid] = a3;
}

// ---------- classifier fc1 ----------
extern "C" __global__ __launch_bounds__(256)
void cls_fc1(const float* __restrict__ pooled, const float* __restrict__ w1,
             const float* __restrict__ b1, float* __restrict__ zpre)
{
  int idx = blockIdx.x * blockDim.x + threadIdx.x;
  if (idx >= BATCH * 128) return;
  int row = idx >> 7, c = idx & 127;
  const float* pr = pooled + (size_t)row * 512;
  float s = b1[c];
  for (int k = 0; k < 512; ++k) s += pr[k] * w1[(size_t)k * 128 + c];
  zpre[idx] = s;
}

// ---------- classifier BN stats over 512 rows (one block per column) ----------
extern "C" __global__ __launch_bounds__(256)
void cls_bn_stats(const float* __restrict__ zpre, const float* __restrict__ g,
                  const float* __restrict__ b, float* __restrict__ scsh)
{
  __shared__ float s1[256], s2[256];
  int c = blockIdx.x, tid = threadIdx.x;
  float v0 = zpre[(size_t)tid * 128 + c];
  float v1 = zpre[(size_t)(tid + 256) * 128 + c];
  s1[tid] = v0 + v1; s2[tid] = v0*v0 + v1*v1;
  __syncthreads();
  for (int off = 128; off >= 1; off >>= 1) {
    if (tid < off) { s1[tid] += s1[tid+off]; s2[tid] += s2[tid+off]; }
    __syncthreads();
  }
  if (tid == 0) {
    float mu = s1[0] / 512.f;
    float var = s2[0] / 512.f - mu * mu;
    float sc = rsqrtf(var + BN_EPS) * g[c];
    scsh[c] = sc;
    scsh[128 + c] = b[c] - mu * sc;
  }
}

// ---------- head: relu(BN) -> logits -> softmax; writes logits then probs ----------
extern "C" __global__ __launch_bounds__(128)
void cls_head(const float* __restrict__ zpre, const float* __restrict__ scsh,
              const float* __restrict__ w2, const float* __restrict__ b2,
              float* __restrict__ out)
{
  __shared__ float z[128];
  __shared__ float lg[20];
  __shared__ float red[2];
  int row = blockIdx.x, tid = threadIdx.x;
  float v = zpre[(size_t)row * 128 + tid] * scsh[tid] + scsh[128 + tid];
  z[tid] = fmaxf(v, 0.f);
  __syncthreads();
  if (tid < 20) {
    float s = b2[tid];
    for (int k = 0; k < 128; ++k) s += z[k] * w2[k * 20 + tid];
    lg[tid] = s;
    out[(size_t)row * 20 + tid] = s;
  }
  __syncthreads();
  if (tid == 0) {
    float mx = lg[0];
    for (int k = 1; k < 20; ++k) mx = fmaxf(mx, lg[k]);
    float s = 0.f;
    for (int k = 0; k < 20; ++k) s += __expf(lg[k] - mx);
    red[0] = mx; red[1] = 1.f / s;
  }
  __syncthreads();
  if (tid < 20)
    out[(size_t)(BATCH * 20) + row * 20 + tid] = __expf(lg[tid] - red[0]) * red[1];
}

// =======================================================================
extern "C" void kernel_launch(void* const* d_in, const int* in_sizes, int n_in,
                              void* d_out, int out_size, void* d_ws, size_t ws_size,
                              hipStream_t stream)
{
  (void)in_sizes; (void)n_in; (void)out_size; (void)ws_size;
  const float* x_seq  = (const float*)d_in[0];
  // d_in[1] = edge_index (unused: fixed skeleton graph handled analytically)
  const float* proj_w = (const float*)d_in[2];
  const float* proj_b = (const float*)d_in[3];
  const float *lw[3], *lb[3], *rw[3], *qw[3], *qb[3], *bg[3], *bb[3];
  int p = 4;
  for (int i = 0; i < 3; ++i) {
    lw[i] = (const float*)d_in[p++]; lb[i] = (const float*)d_in[p++];
    rw[i] = (const float*)d_in[p++]; qw[i] = (const float*)d_in[p++];
    qb[i] = (const float*)d_in[p++]; bg[i] = (const float*)d_in[p++];
    bb[i] = (const float*)d_in[p++];
  }
  const float* attn_w = (const float*)d_in[25];
  const float* attn_b = (const float*)d_in[26];
  const float* cls_w1 = (const float*)d_in[27];
  const float* cls_b1 = (const float*)d_in[28];
  const float* cls_g  = (const float*)d_in[29];
  const float* cls_bb = (const float*)d_in[30];
  const float* cls_w2 = (const float*)d_in[31];
  const float* cls_b2 = (const float*)d_in[32];

  char* base = (char*)d_ws;
  size_t cur = 0;
  auto carve = [&](size_t bytes) -> char* {
    char* r = base + cur;
    cur = (cur + bytes + 255) & ~(size_t)255;
    return r;
  };
  const size_t NNv = (size_t)NTOT;
  unsigned short* xb16 = (unsigned short*)carve(NNv * 256 * 2);
  unsigned short* aggb = (unsigned short*)carve(NNv * 256 * 2);
  float* yA      = (float*)carve(NNv * 256 * 4);
  float* yB      = (float*)carve(NNv * 256 * 4);
  float* xres    = (float*)carve(NNv * 256 * 4);
  float* partial = (float*)carve((size_t)255 * 512 * 4);
  float* stats   = (float*)carve(512 * 4);
  float* pooled  = (float*)carve((size_t)BATCH * 512 * 4);
  float* zpre    = (float*)carve((size_t)BATCH * 128 * 4);
  float* scsh    = (float*)carve(256 * 4);
  int dinA[3]  = {64, 128, 256};
  int doutA[3] = {128, 256, 128};
  unsigned short* wT[3][3];
  for (int i = 0; i < 3; ++i)
    for (int m = 0; m < 3; ++m)
      wT[i][m] = (unsigned short*)carve((size_t)dinA[i] * doutA[i] * 2);

  // 1. weight convert + transpose to bf16
  for (int i = 0; i < 3; ++i) {
    const float* src[3] = {lw[i], rw[i], qw[i]};
    for (int m = 0; m < 3; ++m) {
      int n = dinA[i] * doutA[i];
      wconv_t<<<(n + 255) / 256, 256, 0, stream>>>(src[m], wT[i][m], dinA[i], doutA[i]);
    }
  }
  // 2. input projection -> bf16 node features
  {
    size_t n = NNv * 64;
    proj_kernel<<<(unsigned)((n + 255) / 256), 256, 0, stream>>>(x_seq, proj_w, proj_b, xb16);
  }
  // 3. three SAGE layers
  float* yBuf[3]    = {yA, yB, yA};
  int shiftIn[3]  = {6, 7, 8};
  int shiftOut[3] = {7, 8, 7};
  for (int i = 0; i < 3; ++i) {
    int din = dinA[i], dout = doutA[i];
    size_t tin = NNv << shiftIn[i];
    agg_mean<<<(unsigned)((tin + 255) / 256), 256, 0, stream>>>(xb16, aggb, shiftIn[i]);
    dim3 g(NTOT / 16 / 8, dout / 16);
    sage_gemm<<<g, 256, 0, stream>>>(aggb, xb16, wT[i][0], wT[i][1], wT[i][2],
                                     lb[i], qb[i], yBuf[i], xres, din, dout);
    dim3 gs(dout / 16, 255);
    col_stats_partial<<<gs, 256, 0, stream>>>(yBuf[i], partial, dout);
    col_stats_reduce<<<(2 * dout + 255) / 256, 256, 0, stream>>>(partial, stats, dout, 255);
    size_t tout = NNv << shiftOut[i];
    bn_relu_res<<<(unsigned)((tout + 255) / 256), 256, 0, stream>>>(
        yBuf[i], xres, stats, bg[i], bb[i], xb16, dout, shiftOut[i]);
  }
  // 4. attention pooling (final features live in yA)
  attn_pool<<<BATCH, 128, 0, stream>>>(yA, attn_w, attn_b, pooled);
  // 5. classifier
  cls_fc1<<<(BATCH * 128) / 256, 256, 0, stream>>>(pooled, cls_w1, cls_b1, zpre);
  cls_bn_stats<<<128, 256, 0, stream>>>(zpre, cls_g, cls_bb, scsh);
  cls_head<<<BATCH, 128, 0, stream>>>(zpre, scsh, cls_w2, cls_b2, (float*)d_out);
}
